// AttentivePooling_49048526520634
// MI455X (gfx1250) — matrix operand order repaired and checked by reading, not compiled, not run on hardware
//
#include <hip/hip_runtime.h>
#include <math.h>

#define HID 256
#define HALF_HID 128

typedef __attribute__((ext_vector_type(16))) __bf16 v16bf;
typedef __attribute__((ext_vector_type(8)))  float  v8f;

union Frag {
    v16bf v;
    __bf16 b[16];
    unsigned short u[16];
    unsigned int   d[8];
};

union PkBf {
    __bf16 b[2];
    unsigned int u;
};

#if __has_builtin(__builtin_amdgcn_tanh_f32)
__device__ __forceinline__ float fast_tanh(float x) { return __builtin_amdgcn_tanh_f32(x); }
#elif __has_builtin(__builtin_amdgcn_tanhf)
__device__ __forceinline__ float fast_tanh(float x) { return __builtin_amdgcn_tanhf(x); }
#else
// CDNA5 has native V_TANH_F32 (TRANS op); use it directly.
__device__ __forceinline__ float fast_tanh(float x) {
    float r;
    asm("v_tanh_f32 %0, %1" : "=v"(r) : "v"(x));
    return r;
}
#endif

// ---------------------------------------------------------------------------
// Kernel 1: scores[n] = tanh(x[n,:] @ W1 + b1) @ W2 + b2   via bf16 WMMA
// Block = 256 threads = 8 waves; wave handles 16 rows; block handles 128 rows.
// ---------------------------------------------------------------------------
__global__ __launch_bounds__(256)
void score_kernel(const float* __restrict__ x,
                  const float* __restrict__ W1,
                  const float* __restrict__ b1,
                  const float* __restrict__ W2,
                  const float* __restrict__ b2,
                  float* __restrict__ scores)
{
    // W1 transposed into LDS as bf16: w1t[n][k], n=0..127, k=0..255  (64 KB)
    __shared__ unsigned short w1t[HALF_HID * HID];

    const int tid = threadIdx.x;
    // cooperative load + transpose + pk-convert; one dword store per K-pair.
    // pair p: k = 2*(p/128), n = p%128 ; w1t row-major [n][k] -> dword n*128+k/2
    {
        unsigned int* w1d = (unsigned int*)w1t;
        for (int p = tid; p < HALF_HID * HID / 2; p += 256) {
            int k = (p >> 7) << 1;
            int n = p & 127;
            PkBf pk;
            pk.b[0] = (__bf16)W1[(size_t)k * HALF_HID + n];
            pk.b[1] = (__bf16)W1[(size_t)(k + 1) * HALF_HID + n];
            w1d[n * (HID / 2) + (k >> 1)] = pk.u;
        }
    }
    __syncthreads();

    const int wave = tid >> 5;
    const int lane = tid & 31;
    const int m    = lane & 15;   // row within A tile / col within B,C tiles
    const int hi   = lane >> 4;   // half-wave selector

    const int rowBase = blockIdx.x * 128 + wave * 16;
    const float* xrow = x + (size_t)(rowBase + m) * HID;

    v8f acc[8];
    #pragma unroll
    for (int t = 0; t < 8; ++t) acc[t] = (v8f){0.f,0.f,0.f,0.f,0.f,0.f,0.f,0.f};

    #pragma unroll
    for (int ks = 0; ks < 8; ++ks) {          // K-steps of 32
        const int k0 = ks * 32;

        // ---- A fragment: 16x32 bf16, documented CDNA5 layout ----
        // lanes 0-15: v0..3 -> K=2v,2v+1 ; v4..7 -> K=16+2(v-4)  (hi half: +8)
        // Per lane: two contiguous 8-float runs -> four aligned float4 loads.
        const float4* xr0 = (const float4*)(xrow + k0 + hi * 8);       // K run 0..7
        const float4* xr1 = (const float4*)(xrow + k0 + hi * 8 + 16);  // K run 16..23
        float4 f0 = xr0[0], f1 = xr0[1];
        float4 f2 = xr1[0], f3 = xr1[1];

        Frag a;
        a.b[0]  = (__bf16)f0.x;  a.b[1]  = (__bf16)f0.y;
        a.b[2]  = (__bf16)f0.z;  a.b[3]  = (__bf16)f0.w;
        a.b[4]  = (__bf16)f1.x;  a.b[5]  = (__bf16)f1.y;
        a.b[6]  = (__bf16)f1.z;  a.b[7]  = (__bf16)f1.w;
        a.b[8]  = (__bf16)f2.x;  a.b[9]  = (__bf16)f2.y;
        a.b[10] = (__bf16)f2.z;  a.b[11] = (__bf16)f2.w;
        a.b[12] = (__bf16)f3.x;  a.b[13] = (__bf16)f3.y;
        a.b[14] = (__bf16)f3.z;  a.b[15] = (__bf16)f3.w;

        #pragma unroll
        for (int nt = 0; nt < 8; ++nt) {      // 8 column tiles of 16 -> HID/2=128
            const int n = nt * 16 + m;
            // ---- B fragment: 32x16 bf16; same K order as A: v0..3 -> K=8*hi+2v, v4..7 -> K=16+8*hi+2(v-4)
            // pair (K=kb+2v, kb+2v+1) is one aligned dword in transposed LDS row
            const unsigned int* w1row =
                (const unsigned int*)(w1t + (size_t)n * HID);
            Frag bb;
            const int dbase = (k0 + hi * 8) >> 1;   // dword index
            #pragma unroll
            for (int v = 0; v < 8; ++v) bb.d[v] = w1row[dbase + v + (v >= 4 ? 4 : 0)];

            acc[nt] = __builtin_amdgcn_wmma_f32_16x16x32_bf16(
                          false, a.v, false, bb.v, (short)0, acc[nt],
                          false, false);
        }
    }

    // ---- epilogue: bias + tanh + dot(W2) ----
    // C/D layout: lane holds col n = nt*16 + m, rows M = v + 8*hi
    float partial[8];
    #pragma unroll
    for (int v = 0; v < 8; ++v) partial[v] = 0.f;

    #pragma unroll
    for (int nt = 0; nt < 8; ++nt) {
        const int n = nt * 16 + m;
        const float bias = b1[n];
        const float w2v  = W2[n];
        #pragma unroll
        for (int v = 0; v < 8; ++v)
            partial[v] += fast_tanh(acc[nt][v] + bias) * w2v;
    }

    // reduce over n across the 16 lanes of each half (masks 1,2,4,8 stay in-half)
    #pragma unroll
    for (int off = 1; off < 16; off <<= 1) {
        #pragma unroll
        for (int v = 0; v < 8; ++v)
            partial[v] += __shfl_xor(partial[v], off, 32);
    }

    if (m == 0) {
        const float b2v = b2[0];
        #pragma unroll
        for (int v = 0; v < 8; ++v)
            scores[rowBase + hi * 8 + v] = partial[v] + b2v;
    }
}

// ---------------------------------------------------------------------------
// Kernel 2: per-graph softmax over sorted segments; rewrites w over scores.
// ---------------------------------------------------------------------------
__device__ __forceinline__ int lower_bound(const int* __restrict__ b, int n, int val) {
    int lo = 0, hi = n;
    while (lo < hi) {
        int mid = (lo + hi) >> 1;
        if (b[mid] < val) lo = mid + 1; else hi = mid;
    }
    return lo;
}

__global__ __launch_bounds__(256)
void softmax_kernel(const int* __restrict__ batch, float* __restrict__ sc, int N)
{
    __shared__ float red[256];
    __shared__ int seg[2];
    const int g   = blockIdx.x;
    const int tid = threadIdx.x;

    if (tid == 0) seg[0] = lower_bound(batch, N, g);
    if (tid == 1) seg[1] = lower_bound(batch, N, g + 1);
    __syncthreads();
    const int start = seg[0], end = seg[1];

    // block max
    float mx = -INFINITY;
    for (int i = start + tid; i < end; i += 256) mx = fmaxf(mx, sc[i]);
    red[tid] = mx; __syncthreads();
    for (int s = 128; s > 0; s >>= 1) {
        if (tid < s) red[tid] = fmaxf(red[tid], red[tid + s]);
        __syncthreads();
    }
    const float gmax = red[0];
    __syncthreads();

    // block sum of exp
    float sm = 0.f;
    for (int i = start + tid; i < end; i += 256) sm += __expf(sc[i] - gmax);
    red[tid] = sm; __syncthreads();
    for (int s = 128; s > 0; s >>= 1) {
        if (tid < s) red[tid] += red[tid + s];
        __syncthreads();
    }
    const float inv = 1.0f / red[0];
    __syncthreads();

    for (int i = start + tid; i < end; i += 256)
        sc[i] = __expf(sc[i] - gmax) * inv;
}

// ---------------------------------------------------------------------------
// Kernel 3: pooled[g,c] = sum_{n in segment g} x[n,c] * w[n]
// 256 threads = 256 channels, fully coalesced stream of x.
// ---------------------------------------------------------------------------
__global__ __launch_bounds__(256)
void pool_kernel(const float* __restrict__ x,
                 const int* __restrict__ batch,
                 const float* __restrict__ w,
                 float* __restrict__ out, int N)
{
    __shared__ int seg[2];
    const int g = blockIdx.x;
    const int c = threadIdx.x;

    if (c == 0) seg[0] = lower_bound(batch, N, g);
    if (c == 1) seg[1] = lower_bound(batch, N, g + 1);
    __syncthreads();
    const int start = seg[0], end = seg[1];

    float acc = 0.f;
    for (int n = start; n < end; ++n) {
        if (n + 8 < end)
            __builtin_prefetch(x + (size_t)(n + 8) * HID + c, 0, 1);
        acc += x[(size_t)n * HID + c] * w[n];
    }
    out[(size_t)g * HID + c] = acc;
}

// ---------------------------------------------------------------------------
extern "C" void kernel_launch(void* const* d_in, const int* in_sizes, int n_in,
                              void* d_out, int out_size, void* d_ws, size_t ws_size,
                              hipStream_t stream)
{
    const float* x     = (const float*)d_in[0];
    const int*   batch = (const int*)  d_in[1];
    const float* W1    = (const float*)d_in[2];
    const float* b1    = (const float*)d_in[3];
    const float* W2    = (const float*)d_in[4];
    const float* b2    = (const float*)d_in[5];
    float*       out   = (float*)d_out;

    const int N = in_sizes[1];           // number of nodes
    const int G = out_size / HID;        // number of graphs

    float* scores = (float*)d_ws;        // N floats; rewritten in place as w

    score_kernel<<<N / 128, 256, 0, stream>>>(x, W1, b1, W2, b2, scores);
    softmax_kernel<<<G, 256, 0, stream>>>(batch, scores, N);
    pool_kernel<<<G, 256, 0, stream>>>(x, batch, scores, out, N);
}
